// SpectralConv2d_15075335209171
// MI455X (gfx1250) — matrix operand order repaired and checked by reading, not compiled, not run on hardware
//
#include <hip/hip_runtime.h>

// ---------------------------------------------------------------------------
// SpectralConv2d on MI455X (gfx1250), factorized into WMMA f16 GEMMs.
//   x:        [8, 32, 256, 256] f32
//   weights1: [32, 32, 32, 32]  f32   (modes rows 0..31)
//   weights2: [32, 32, 32, 32]  f32   (modes rows 224..255)
//   y:        [8, 32, 256, 256] f32
//
// Pipeline (per image = batch*channel):
//   A: CS(512x32) = x(256x256) @ [cos|sin](256x64)          K=256  (f16 WMMA)
//   B: X(64x32)   = [cosU|-sinU](64x512) @ CS               K=512  (f16 WMMA)
//   C: Y[b,o,m]   = 2^-8 * sum_i X[b,i,m]*w[i,o,m]          (f32 VALU)
//   D1: PQ(128x256) = [Y*cosV ; Y*sinV]                     K=32   (f16 WMMA)
//   D2: y(256x256)  = [cosU|-sinU]^T*2^-8 (256x128) @ PQ    K=128  (f16 WMMA)
// Multi-accumulator waves reuse the shared fragment to minimize HBM traffic.
// ---------------------------------------------------------------------------

typedef _Float16 h16;
typedef __attribute__((ext_vector_type(16))) _Float16 v16h;
typedef __attribute__((ext_vector_type(8)))  _Float16 v8h;
typedef __attribute__((ext_vector_type(8)))  float    v8f;

#define NIMG   256      // 8 batch * 32 channels
#define TWO_PI 6.28318530717958647692f

// ---- workspace layout (bytes) ----------------------------------------------
#define B_B1T   (size_t)0               // f16 [64][256]  B1 N-major (cos|sin along w)
#define B_A2    (size_t)32768           // f16 [64][512]  A2 row-major ([cosU|-sinU])
#define B_D1BT  (size_t)98304           // f16 [256][64]  D1 basis N-major (cos|sin)
#define B_A4    (size_t)131072          // f16 [256][128] A4 row-major (scaled 2^-8)
#define B_CST   (size_t)196608          // f16 [256 img][32][512]  CS, N-major
#define B_X     (size_t)8585216         // f32 [256 img][64][32]   Re(FFT2) modes
#define B_Y     (size_t)10682368        // f16 [256 img][64][32]   mixed modes
#define B_PQT   (size_t)11730944        // f16 [256 img][256][128] PQ, N-major
// total = 28,508,160 bytes

// ---------------------------------------------------------------------------
// Fragment helpers — layouts per CDNA5 ISA 7.12.2 (wave32).
// A 16x32 f16: lane m = L%16 (row), half = L/16; element e holds
//   k = (e<8 ? e : 8+e) + 8*half  ->  two contiguous runs of 8 halves.
// B 32x16 f16 (N-major storage [n][k]): lane n = L%16, half = L/16;
//   element e holds k = (e<8 ? e : 8+e) + 8*half (same as A).
// C/D f32: lane n = L%16; VGPR r holds row m = r + 8*(L/16).
// ---------------------------------------------------------------------------

__device__ __forceinline__ v16h load_a_f16(const h16* __restrict__ base, int ld,
                                           int row0, int k0, int lane) {
  int m = lane & 15, hf = lane >> 4;
  const h16* p = base + (size_t)(row0 + m) * ld + k0 + 8 * hf;
  v8h lo = *(const v8h*)(p);
  v8h hi = *(const v8h*)(p + 16);
  v16h a;
#pragma unroll
  for (int e = 0; e < 8; ++e) { a[e] = lo[e]; a[e + 8] = hi[e]; }
  return a;
}

__device__ __forceinline__ v16h load_a_f32cvt(const float* __restrict__ base, int ld,
                                              int row0, int k0, int lane) {
  int m = lane & 15, hf = lane >> 4;
  const float* p = base + (size_t)(row0 + m) * ld + k0 + 8 * hf;
  v16h a;
#pragma unroll
  for (int e = 0; e < 8; ++e) {
    a[e]     = (h16)p[e];
    a[e + 8] = (h16)p[e + 16];
  }
  return a;
}

__device__ __forceinline__ v16h load_b_nmajor(const h16* __restrict__ base, int ldk,
                                              int n0, int k0, int lane) {
  int n = lane & 15, hf = lane >> 4;
  const h16* p = base + (size_t)(n0 + n) * ldk + k0 + 8 * hf;
  v8h lo = *(const v8h*)(p);
  v8h hi = *(const v8h*)(p + 16);
  v16h b;
#pragma unroll
  for (int e = 0; e < 8; ++e) { b[e] = lo[e]; b[e + 8] = hi[e]; }
  return b;
}

__device__ __forceinline__ v8f wmma_f16(v16h a, v16h b, v8f c) {
  return __builtin_amdgcn_wmma_f32_16x16x32_f16(false, a, false, b,
                                                (short)0, c, false, false);
}

// ---------------------------------------------------------------------------
// Kernel 0: build DFT basis matrices (f16) in workspace. Runs every call.
// ---------------------------------------------------------------------------
__global__ void init_basis(h16* __restrict__ B1T, h16* __restrict__ A2,
                           h16* __restrict__ D1BT, h16* __restrict__ A4) {
  int idx = blockIdx.x * blockDim.x + threadIdx.x;
  const float s = TWO_PI / 256.0f;
  if (idx < 64 * 256) {                       // B1T[v64][w]
    int v = idx >> 8, w = idx & 255;
    int vm = v & 31;
    float ang = (float)((vm * w) & 255) * s;
    B1T[idx] = (h16)((v < 32) ? cosf(ang) : sinf(ang));
    return;
  }
  idx -= 64 * 256;
  if (idx < 64 * 512) {                       // A2[r][k] = [cosU | -sinU]
    int r = idx >> 9, k = idx & 511;
    int u = (r < 32) ? r : (192 + r);         // 224..255 for r=32..63
    int kk = k & 255;
    float ang = (float)((u * kk) & 255) * s;
    A2[idx] = (h16)((k < 256) ? cosf(ang) : -sinf(ang));
    return;
  }
  idx -= 64 * 512;
  if (idx < 256 * 64) {                       // D1BT[w][c] = [cosV | sinV]
    int w = idx >> 6, c = idx & 63;
    int cm = c & 31;
    float ang = (float)((cm * w) & 255) * s;
    D1BT[idx] = (h16)((c < 32) ? cosf(ang) : sinf(ang));
    return;
  }
  idx -= 256 * 64;
  if (idx < 256 * 128) {                      // A4[h][c], scaled by 2^-8
    int h = idx >> 7, c = idx & 127;
    int cm = c & 63;
    int u = (cm < 32) ? cm : (192 + cm);
    float ang = (float)((u * h) & 255) * s;
    float v = (c < 64) ? cosf(ang) : -sinf(ang);
    A4[idx] = (h16)(v * (1.0f / 256.0f));
  }
}

// ---------------------------------------------------------------------------
// Stage A: CS = x @ B1 (K=256). One wave covers the full N=64: the x
// A-fragment (the dominant HBM stream) is loaded ONCE and fed to 4
// accumulators. grid (16, 256), block 32 (1 wave).
// Output N-major f16 [img][32][512]: cols 0..31 -> cos at k=h,
// cols 32..63 -> sin at k=256+h.
// ---------------------------------------------------------------------------
__global__ void dht_rows(const float* __restrict__ x,
                         const h16* __restrict__ B1T,
                         h16* __restrict__ CST) {
  int lane = threadIdx.x;
  int mtile = blockIdx.x, img = blockIdx.y;
  const float* xp = x + (size_t)img * 65536;
  v8f acc[4];
#pragma unroll
  for (int nt = 0; nt < 4; ++nt)
    acc[nt] = (v8f){0.f, 0.f, 0.f, 0.f, 0.f, 0.f, 0.f, 0.f};
#pragma unroll
  for (int kk = 0; kk < 8; ++kk) {
    v16h a = load_a_f32cvt(xp, 256, mtile * 16, kk * 32, lane);
    if (kk < 7)   // cover latency of the next cold x K-block
      __builtin_prefetch(xp + (size_t)(mtile * 16 + (lane & 15)) * 256
                            + (kk + 1) * 32, 0, 1);
#pragma unroll
    for (int nt = 0; nt < 4; ++nt) {
      v16h b = load_b_nmajor(B1T, 256, nt * 16, kk * 32, lane);
      acc[nt] = wmma_f16(a, b, acc[nt]);
    }
  }
  int n = lane & 15, hf = lane >> 4;
#pragma unroll
  for (int nt = 0; nt < 4; ++nt) {
    int v64 = nt * 16 + n;
    int kbase = ((v64 >= 32) ? 256 : 0) + mtile * 16 + 8 * hf;
    h16* dst = CST + (size_t)img * 16384 + (size_t)(v64 & 31) * 512 + kbase;
#pragma unroll
    for (int r = 0; r < 8; ++r) dst[r] = (h16)acc[nt][r];
  }
}

// ---------------------------------------------------------------------------
// Stage B: X = A2(64x512) @ CS(512x32). One wave covers N=32 (2 accumulators,
// A2 fragment loaded once). grid (4, 256), block 32.
// ---------------------------------------------------------------------------
__global__ void dht_cols(const h16* __restrict__ A2,
                         const h16* __restrict__ CST,
                         float* __restrict__ X) {
  int lane = threadIdx.x;
  int rtile = blockIdx.x, img = blockIdx.y;
  const h16* cs = CST + (size_t)img * 16384;
  v8f acc[2];
  acc[0] = (v8f){0.f, 0.f, 0.f, 0.f, 0.f, 0.f, 0.f, 0.f};
  acc[1] = acc[0];
#pragma unroll
  for (int kk = 0; kk < 16; ++kk) {
    v16h a = load_a_f16(A2, 512, rtile * 16, kk * 32, lane);
#pragma unroll
    for (int vt = 0; vt < 2; ++vt) {
      v16h b = load_b_nmajor(cs, 512, vt * 16, kk * 32, lane);
      acc[vt] = wmma_f16(a, b, acc[vt]);
    }
  }
  int n = lane & 15, hf = lane >> 4;
  float* out = X + (size_t)img * 2048;
#pragma unroll
  for (int vt = 0; vt < 2; ++vt)
#pragma unroll
    for (int r = 0; r < 8; ++r)
      out[(size_t)(rtile * 16 + r + 8 * hf) * 32 + vt * 16 + n] = acc[vt][r];
}

// ---------------------------------------------------------------------------
// Stage C: per-mode channel mix (f32 weights, f32 accumulate):
//   Y[b,o,r,v] = (1/256) * sum_i X[b,i,r,v] * w{1|2}[i,o,r',v]
// 524288 threads, 32 FMAs each.
// ---------------------------------------------------------------------------
__global__ void mode_mix(const float* __restrict__ X,
                         const float* __restrict__ w1,
                         const float* __restrict__ w2,
                         h16* __restrict__ Y) {
  int t = blockIdx.x * blockDim.x + threadIdx.x;
  int v = t & 31;
  int r = (t >> 5) & 63;
  int o = (t >> 11) & 31;
  int b = t >> 16;
  const float* xp = X + (size_t)(b * 32) * 2048 + r * 32 + v;
  const float* wp = (r < 32)
      ? (w1 + ((size_t)o * 32 + r) * 32 + v)
      : (w2 + ((size_t)o * 32 + (r - 32)) * 32 + v);
  float acc = 0.f;
#pragma unroll
  for (int i = 0; i < 32; ++i)
    acc += xp[(size_t)i * 2048] * wp[(size_t)i * 32768];
  Y[((size_t)(b * 32 + o)) * 2048 + r * 32 + v] = (h16)(acc * (1.0f / 256.0f));
}

// ---------------------------------------------------------------------------
// Stage D1: PQ(128x256) = [Y*cosV ; Y*sinV], K=32 (one WMMA per tile).
// One wave covers 4 w-tiles (Y A-fragment loaded once, 4 accumulators).
// Output N-major f16 [img][256][128]. grid (4, 8, 256), block 32.
// ---------------------------------------------------------------------------
__global__ void imodes(const h16* __restrict__ Y,
                       const h16* __restrict__ D1BT,
                       h16* __restrict__ PQT) {
  int lane = threadIdx.x;
  int wgrp = blockIdx.x, ctile = blockIdx.y, img = blockIdx.z;
  const h16* yp = Y + (size_t)img * 2048;
  v16h a = load_a_f16(yp, 32, (ctile & 3) * 16, 0, lane);
  int koff = (ctile < 4) ? 0 : 32;
  int n = lane & 15, hf = lane >> 4;
#pragma unroll
  for (int j = 0; j < 4; ++j) {
    int wtile = wgrp * 4 + j;
    v16h b = load_b_nmajor(D1BT, 64, wtile * 16, koff, lane);
    v8f acc = {0.f, 0.f, 0.f, 0.f, 0.f, 0.f, 0.f, 0.f};
    acc = wmma_f16(a, b, acc);
    h16* dst = PQT + (size_t)img * 32768 + (size_t)(wtile * 16 + n) * 128
             + ctile * 16 + 8 * hf;
#pragma unroll
    for (int r = 0; r < 8; ++r) dst[r] = (h16)acc[r];
  }
}

// ---------------------------------------------------------------------------
// Stage D2: y = A4(256x128) @ PQ(128x256), f32 out. One wave covers 2 h-tiles
// (PQ B-fragment loaded once, halving L2 re-reads). grid (16, 8, 256).
// ---------------------------------------------------------------------------
__global__ void idht(const h16* __restrict__ A4,
                     const h16* __restrict__ PQT,
                     float* __restrict__ out) {
  int lane = threadIdx.x;
  int wtile = blockIdx.x, hgrp = blockIdx.y, img = blockIdx.z;
  const h16* pq = PQT + (size_t)img * 32768;
  v8f acc[2];
  acc[0] = (v8f){0.f, 0.f, 0.f, 0.f, 0.f, 0.f, 0.f, 0.f};
  acc[1] = acc[0];
#pragma unroll
  for (int kk = 0; kk < 4; ++kk) {
    v16h b = load_b_nmajor(pq, 128, wtile * 16, kk * 32, lane);
#pragma unroll
    for (int j = 0; j < 2; ++j) {
      v16h a = load_a_f16(A4, 128, (hgrp * 2 + j) * 16, kk * 32, lane);
      acc[j] = wmma_f16(a, b, acc[j]);
    }
  }
  int n = lane & 15, hf = lane >> 4;
  float* op = out + (size_t)img * 65536;
#pragma unroll
  for (int j = 0; j < 2; ++j)
#pragma unroll
    for (int r = 0; r < 8; ++r)
      op[(size_t)((hgrp * 2 + j) * 16 + r + 8 * hf) * 256 + wtile * 16 + n]
          = acc[j][r];
}

// ---------------------------------------------------------------------------
extern "C" void kernel_launch(void* const* d_in, const int* in_sizes, int n_in,
                              void* d_out, int out_size, void* d_ws, size_t ws_size,
                              hipStream_t stream) {
  (void)in_sizes; (void)n_in; (void)out_size; (void)ws_size;
  const float* x  = (const float*)d_in[0];
  const float* w1 = (const float*)d_in[1];
  const float* w2 = (const float*)d_in[2];
  float* y = (float*)d_out;
  char* ws = (char*)d_ws;

  h16*   B1T  = (h16*)(ws + B_B1T);
  h16*   A2   = (h16*)(ws + B_A2);
  h16*   D1BT = (h16*)(ws + B_D1BT);
  h16*   A4   = (h16*)(ws + B_A4);
  h16*   CST  = (h16*)(ws + B_CST);
  float* X    = (float*)(ws + B_X);
  h16*   Y    = (h16*)(ws + B_Y);
  h16*   PQT  = (h16*)(ws + B_PQT);

  init_basis<<<384, 256, 0, stream>>>(B1T, A2, D1BT, A4);
  dht_rows<<<dim3(16, NIMG), 32, 0, stream>>>(x, B1T, CST);
  dht_cols<<<dim3(4, NIMG), 32, 0, stream>>>(A2, CST, X);
  mode_mix<<<2048, 256, 0, stream>>>(X, w1, w2, Y);
  imodes<<<dim3(4, 8, NIMG), 32, 0, stream>>>(Y, D1BT, PQT);
  idht<<<dim3(16, 8, NIMG), 32, 0, stream>>>(A4, PQT, y);
}
